// GATNet_86887188398787
// MI455X (gfx1250) — compile-verified, matrix operand order checked by reading, not run on hardware
//
#include <hip/hip_runtime.h>
#include <stdint.h>

// ---------------------------------------------------------------------------
// Types for CDNA5 WMMA
// ---------------------------------------------------------------------------
typedef __attribute__((ext_vector_type(16))) __bf16 v16bf;
typedef __attribute__((ext_vector_type(8)))  float  v8f;

union Frag16 { uint4 q[2]; v16bf v; };

__device__ __forceinline__ unsigned pack2bf(float a, float b) {
    unsigned short ua = __builtin_bit_cast(unsigned short, (__bf16)a);
    unsigned short ub = __builtin_bit_cast(unsigned short, (__bf16)b);
    return (unsigned)ua | ((unsigned)ub << 16);
}

// ---------------------------------------------------------------------------
// GEMM: Hout[N,Fout] = A[N,K] @ W[Fout,K]^T   (bf16 WMMA, f32 accumulate)
// Workgroup tile 128x128, 8 waves (2 M x 4 N), wave tile 64x32, K-step 32.
// K must be a multiple of 32 (true for all layers: 256/3584/3072/2048).
// ---------------------------------------------------------------------------
__global__ __launch_bounds__(256)
void gat_gemm_bf16(const float* __restrict__ A, const float* __restrict__ W,
                   float* __restrict__ Hout, int Nrows, int K, int Fout) {
    __shared__ __bf16 As[128 * 40];   // 32 cols + 8 pad
    __shared__ __bf16 Bs[128 * 40];

    const int tid  = threadIdx.x;
    const int lane = tid & 31;
    const int wave = tid >> 5;
    const int wm   = (wave >> 2) * 64;   // 0 or 64
    const int wn   = (wave & 3) * 32;    // 0,32,64,96
    const int blockM = blockIdx.y * 128;
    const int blockN = blockIdx.x * 128;

    v8f acc[4][2];
    #pragma unroll
    for (int i = 0; i < 4; ++i)
        #pragma unroll
        for (int j = 0; j < 2; ++j)
            #pragma unroll
            for (int k = 0; k < 8; ++k) acc[i][j][k] = 0.0f;

    const int lrow = tid >> 1;          // 0..127
    const int lcol = (tid & 1) * 16;    // 0 or 16

    for (int kb = 0; kb < K; kb += 32) {
        // ---- stage A tile (rows of activations) as bf16 ----
        {
            int rg = blockM + lrow;
            unsigned* dst = (unsigned*)&As[lrow * 40 + lcol];
            if (rg < Nrows) {
                const float* ap = A + (size_t)rg * K + kb + lcol;
                #pragma unroll
                for (int j = 0; j < 4; ++j) {
                    float4 v = *(const float4*)(ap + j * 4);
                    dst[j * 2 + 0] = pack2bf(v.x, v.y);
                    dst[j * 2 + 1] = pack2bf(v.z, v.w);
                }
            } else {
                #pragma unroll
                for (int j = 0; j < 8; ++j) dst[j] = 0u;
            }
        }
        // ---- stage B tile (rows of W = output features) as bf16 ----
        {
            int ng = blockN + lrow;
            unsigned* dst = (unsigned*)&Bs[lrow * 40 + lcol];
            if (ng < Fout) {
                const float* bp = W + (size_t)ng * K + kb + lcol;
                #pragma unroll
                for (int j = 0; j < 4; ++j) {
                    float4 v = *(const float4*)(bp + j * 4);
                    dst[j * 2 + 0] = pack2bf(v.x, v.y);
                    dst[j * 2 + 1] = pack2bf(v.z, v.w);
                }
            } else {
                #pragma unroll
                for (int j = 0; j < 8; ++j) dst[j] = 0u;
            }
        }
        __syncthreads();

        // ---- load fragments per the 16-bit 16x32 A-layout ----
        Frag16 afrag[4], bfrag[2];
        const int fr = lane & 15;
        const int fk = (lane >> 4) * 8;   // 0 or 8
        #pragma unroll
        for (int mf = 0; mf < 4; ++mf) {
            const __bf16* p = &As[(wm + mf * 16 + fr) * 40];
            afrag[mf].q[0] = *(const uint4*)(p + fk);
            afrag[mf].q[1] = *(const uint4*)(p + 16 + fk);
        }
        #pragma unroll
        for (int nf = 0; nf < 2; ++nf) {
            const __bf16* p = &Bs[(wn + nf * 16 + fr) * 40];
            bfrag[nf].q[0] = *(const uint4*)(p + fk);
            bfrag[nf].q[1] = *(const uint4*)(p + 16 + fk);
        }
        #pragma unroll
        for (int mf = 0; mf < 4; ++mf)
            #pragma unroll
            for (int nf = 0; nf < 2; ++nf)
                acc[mf][nf] = __builtin_amdgcn_wmma_f32_16x16x32_bf16(
                    false, afrag[mf].v, false, bfrag[nf].v,
                    (short)0, acc[mf][nf], false, false);
        __syncthreads();
    }

    // ---- store: C/D layout — VGPR r: lanes 0-15 M=r, lanes 16-31 M=r+8 ----
    const int sm = (lane >> 4) * 8;
    const int sn = lane & 15;
    #pragma unroll
    for (int mf = 0; mf < 4; ++mf)
        #pragma unroll
        for (int nf = 0; nf < 2; ++nf) {
            int n = blockN + wn + nf * 16 + sn;
            if (n < Fout) {
                #pragma unroll
                for (int r = 0; r < 8; ++r) {
                    int m = blockM + wm + mf * 16 + sm + r;
                    if (m < Nrows) Hout[(size_t)m * Fout + n] = acc[mf][nf][r];
                }
            }
        }
}

// ---------------------------------------------------------------------------
// CSR construction (edges fixed for all 4 layers)
// ---------------------------------------------------------------------------
__global__ void build_edges(const int* __restrict__ edge_index,
                            int* __restrict__ srcA, int* __restrict__ dstA,
                            int* __restrict__ cnt, int E, int Nn) {
    int e = blockIdx.x * blockDim.x + threadIdx.x;
    int ET = E + Nn;
    if (e >= ET) return;
    int s, d;
    if (e < E) { s = edge_index[e]; d = edge_index[E + e]; }
    else       { s = d = e - E; }                 // self loops appended
    srcA[e] = s;
    dstA[e] = d;
    atomicAdd(&cnt[d], 1);
}

__global__ __launch_bounds__(1024)
void csr_scan(const int* __restrict__ cnt, int* __restrict__ row_start,
              int* __restrict__ cursor, int n) {
    __shared__ int sh[1024];
    int tid = threadIdx.x;
    int chunk = (n + 1023) / 1024;
    int base = tid * chunk;
    int local = 0;
    for (int i = 0; i < chunk; ++i) { int idx = base + i; if (idx < n) local += cnt[idx]; }
    sh[tid] = local;
    __syncthreads();
    for (int off = 1; off < 1024; off <<= 1) {
        int v = (tid >= off) ? sh[tid - off] : 0;
        __syncthreads();
        sh[tid] += v;
        __syncthreads();
    }
    int run = sh[tid] - local;   // exclusive prefix
    for (int i = 0; i < chunk; ++i) {
        int idx = base + i;
        if (idx < n) { row_start[idx] = run; cursor[idx] = run; run += cnt[idx]; }
    }
    if (tid == 1023) row_start[n] = sh[1023];
}

__global__ void csr_scatter(const int* __restrict__ dstA, int* __restrict__ cursor,
                            int* __restrict__ eidx, int ET) {
    int e = blockIdx.x * blockDim.x + threadIdx.x;
    if (e >= ET) return;
    int pos = atomicAdd(&cursor[dstA[e]], 1);
    eidx[pos] = e;
}

// ---------------------------------------------------------------------------
// Attention logits: al_s[n,h] = <h[n,h,:], a_src[h,:]>, same for a_dst
// ---------------------------------------------------------------------------
__global__ void attn_logits(const float* __restrict__ H,
                            const float* __restrict__ a_src,
                            const float* __restrict__ a_dst,
                            float* __restrict__ al_s, float* __restrict__ al_d,
                            int Nn, int Hh, int C) {
    int t = blockIdx.x * blockDim.x + threadIdx.x;
    if (t >= Nn * Hh) return;
    int n = t / Hh, h = t % Hh;
    const float* hp = H + (size_t)n * Hh * C + (size_t)h * C;
    const float* as = a_src + (size_t)h * C;
    const float* ad = a_dst + (size_t)h * C;
    float ss = 0.f, sd = 0.f;
    for (int c = 0; c < C; ++c) { float v = hp[c]; ss += v * as[c]; sd += v * ad[c]; }
    al_s[t] = ss;
    al_d[t] = sd;
}

// ---------------------------------------------------------------------------
// Segment softmax over incoming edges of each destination node
// ---------------------------------------------------------------------------
__global__ void edge_softmax(const float* __restrict__ al_s,
                             const float* __restrict__ al_d,
                             const int* __restrict__ row_start,
                             const int* __restrict__ eidx,
                             const int* __restrict__ srcA,
                             float* __restrict__ alpha, int Nn, int Hh) {
    int t = blockIdx.x * blockDim.x + threadIdx.x;
    if (t >= Nn * Hh) return;
    int n = t / Hh, h = t % Hh;
    int rs = row_start[n], re = row_start[n + 1];
    float ald = al_d[n * Hh + h];
    float m = -1e30f;
    for (int i = rs; i < re; ++i) {
        int e = eidx[i];
        float v = al_s[srcA[e] * Hh + h] + ald;
        v = (v >= 0.f) ? v : 0.2f * v;
        m = fmaxf(m, v);
    }
    float s = 0.f;
    for (int i = rs; i < re; ++i) {
        int e = eidx[i];
        float v = al_s[srcA[e] * Hh + h] + ald;
        v = (v >= 0.f) ? v : 0.2f * v;
        float ex = __expf(v - m);
        alpha[(size_t)e * Hh + h] = ex;
        s += ex;
    }
    float inv = 1.0f / (s + 1e-16f);
    for (int i = rs; i < re; ++i) {
        int e = eidx[i];
        alpha[(size_t)e * Hh + h] *= inv;
    }
}

// ---------------------------------------------------------------------------
// Aggregation: out[d,f] = sum_e alpha[e, f/C] * H[src[e], f]  (+bias, relu)
// One block per destination node; per-thread register accumulators.
// ---------------------------------------------------------------------------
__global__ __launch_bounds__(256)
void aggregate(const float* __restrict__ Hbuf, const float* __restrict__ alpha,
               const int* __restrict__ row_start, const int* __restrict__ eidx,
               const int* __restrict__ srcA, const float* __restrict__ bias,
               float* __restrict__ out, int Hh, int C, int do_relu) {
    const int d = blockIdx.x;
    const int F = Hh * C;
    const int tid = threadIdx.x;
    const int rs = row_start[d], re = row_start[d + 1];
    const int nf = (F + 255) / 256;   // <= 14

    float acc[14];
    int   hd[14];
    for (int j = 0; j < nf; ++j) {
        int f = tid + j * 256;
        acc[j] = 0.f;
        hd[j] = (f < F) ? (f / C) : 0;
    }
    for (int i = rs; i < re; ++i) {
        int e = eidx[i];
        const float* hrow = Hbuf + (size_t)srcA[e] * F;
        const float* arow = alpha + (size_t)e * Hh;
        for (int j = 0; j < nf; ++j) {
            int f = tid + j * 256;
            if (f < F) acc[j] += arow[hd[j]] * hrow[f];
        }
    }
    for (int j = 0; j < nf; ++j) {
        int f = tid + j * 256;
        if (f < F) {
            float v = acc[j] + bias[f];
            if (do_relu) v = fmaxf(v, 0.f);
            out[(size_t)d * F + f] = v;
        }
    }
}

// ---------------------------------------------------------------------------
// Graph LayerNorm: scalar mean/var over the whole [N,F] matrix
// ---------------------------------------------------------------------------
__global__ __launch_bounds__(256)
void ln_reduce(const float* __restrict__ x, double* __restrict__ stats, long long n) {
    __shared__ double s1[256], s2[256];
    double a = 0.0, b = 0.0;
    for (long long i = (long long)blockIdx.x * blockDim.x + threadIdx.x; i < n;
         i += (long long)gridDim.x * blockDim.x) {
        double v = (double)x[i];
        a += v; b += v * v;
    }
    s1[threadIdx.x] = a; s2[threadIdx.x] = b;
    __syncthreads();
    for (int st = 128; st > 0; st >>= 1) {
        if (threadIdx.x < st) { s1[threadIdx.x] += s1[threadIdx.x + st];
                                s2[threadIdx.x] += s2[threadIdx.x + st]; }
        __syncthreads();
    }
    if (threadIdx.x == 0) {
        atomicAdd(&stats[0], s1[0]);
        atomicAdd(&stats[1], s2[0]);
    }
}

__global__ void ln_apply(float* __restrict__ x, const float* __restrict__ w,
                         const float* __restrict__ b, const double* __restrict__ stats,
                         long long n, int F) {
    long long i = (long long)blockIdx.x * blockDim.x + threadIdx.x;
    if (i >= n) return;
    double mu  = stats[0] / (double)n;
    double var = stats[1] / (double)n - mu * mu;
    float rs = (float)(1.0 / sqrt(var + 1e-5));
    int f = (int)(i % F);
    x[i] = (float)((double)x[i] - mu) * rs * w[f] + b[f];
}

// ---------------------------------------------------------------------------
// Host driver
// ---------------------------------------------------------------------------
extern "C" void kernel_launch(void* const* d_in, const int* in_sizes, int n_in,
                              void* d_out, int out_size, void* d_ws, size_t ws_size,
                              hipStream_t stream) {
    (void)n_in; (void)out_size; (void)ws_size;
    const float* x          = (const float*)d_in[0];
    const int*   edge_index = (const int*)d_in[1];
    const float* W[4]    = {(const float*)d_in[2],  (const float*)d_in[8],
                            (const float*)d_in[14], (const float*)d_in[20]};
    const float* asrc[4] = {(const float*)d_in[3],  (const float*)d_in[9],
                            (const float*)d_in[15], (const float*)d_in[21]};
    const float* adst[4] = {(const float*)d_in[4],  (const float*)d_in[10],
                            (const float*)d_in[16], (const float*)d_in[22]};
    const float* bias[4] = {(const float*)d_in[5],  (const float*)d_in[11],
                            (const float*)d_in[17], (const float*)d_in[23]};
    const float* lnw[3]  = {(const float*)d_in[6],  (const float*)d_in[12],
                            (const float*)d_in[18]};
    const float* lnb[3]  = {(const float*)d_in[7],  (const float*)d_in[13],
                            (const float*)d_in[19]};

    const int N  = in_sizes[0] / 256;   // 10000
    const int E  = in_sizes[1] / 2;     // 50000
    const int ET = E + N;               // + self loops

    const int fin[4]   = {256, 3584, 3072, 2048};
    const int fout[4]  = {3584, 3072, 2048, 1000};
    const int heads[4] = {8, 8, 8, 1};
    const int cdim[4]  = {448, 384, 256, 1000};

    // ---- workspace carve (256B aligned) ----
    char* ws = (char*)d_ws;
    size_t off = 0;
    auto carve = [&](size_t bytes) -> char* {
        char* p = ws + off;
        off = (off + bytes + 255) & ~(size_t)255;
        return p;
    };
    float*  bufA      = (float*)carve((size_t)N * 3584 * 4);
    float*  bufH      = (float*)carve((size_t)N * 3584 * 4);
    float*  al_s      = (float*)carve((size_t)N * 8 * 4);
    float*  al_d      = (float*)carve((size_t)N * 8 * 4);
    float*  alpha     = (float*)carve((size_t)ET * 8 * 4);
    int*    srcA      = (int*)carve((size_t)ET * 4);
    int*    dstA      = (int*)carve((size_t)ET * 4);
    int*    eidx      = (int*)carve((size_t)ET * 4);
    int*    cnt       = (int*)carve((size_t)N * 4);
    int*    row_start = (int*)carve((size_t)(N + 1) * 4);
    int*    cursor    = (int*)carve((size_t)N * 4);
    double* stats     = (double*)carve(16);

    // ---- CSR (once; same edges every layer) ----
    hipMemsetAsync(cnt, 0, (size_t)N * 4, stream);
    build_edges<<<(ET + 255) / 256, 256, 0, stream>>>(edge_index, srcA, dstA, cnt, E, N);
    csr_scan<<<1, 1024, 0, stream>>>(cnt, row_start, cursor, N);
    csr_scatter<<<(ET + 255) / 256, 256, 0, stream>>>(dstA, cursor, eidx, ET);

    for (int li = 0; li < 4; ++li) {
        const float* Ain = (li == 0) ? x : bufA;
        float* Out = (li == 3) ? (float*)d_out : bufA;
        const int Fo = fout[li];

        dim3 gg((Fo + 127) / 128, (N + 127) / 128);
        gat_gemm_bf16<<<gg, 256, 0, stream>>>(Ain, W[li], bufH, N, fin[li], Fo);

        int nh = N * heads[li];
        attn_logits<<<(nh + 255) / 256, 256, 0, stream>>>(
            bufH, asrc[li], adst[li], al_s, al_d, N, heads[li], cdim[li]);
        edge_softmax<<<(nh + 255) / 256, 256, 0, stream>>>(
            al_s, al_d, row_start, eidx, srcA, alpha, N, heads[li]);
        aggregate<<<N, 256, 0, stream>>>(
            bufH, alpha, row_start, eidx, srcA, bias[li], Out,
            heads[li], cdim[li], (li < 3) ? 1 : 0);

        if (li < 3) {
            hipMemsetAsync(stats, 0, 16, stream);
            long long tot = (long long)N * Fo;
            ln_reduce<<<1024, 256, 0, stream>>>(Out, stats, tot);
            ln_apply<<<(int)((tot + 255) / 256), 256, 0, stream>>>(
                Out, lnw[li], lnb[li], stats, tot, Fo);
        }
    }
}
